// SpatioSpectralMambaBlock_18992345383658
// MI455X (gfx1250) — compile-verified, matrix operand order checked by reading, not run on hardware
//
#include <hip/hip_runtime.h>
#include <hip/hip_bf16.h>
#include <math.h>
#include <stdint.h>

// ---------------------------------------------------------------------------
// SpatioSpectralMambaBlock for MI455X (gfx1250, wave32, WMMA).
// Heavy GEMMs (attention qkv/out) -> bf16 WMMA fed by async global->LDS DMA
// (GLOBAL_LOAD_ASYNC_TO_LDS_B128 + s_wait_asynccnt, double buffered).
// Mamba projections -> direct-load bf16 WMMA. Scans/convs/softmax -> VALU.
// ---------------------------------------------------------------------------

typedef __bf16 bf16_t;
typedef __attribute__((ext_vector_type(16))) __bf16 v16bf;
typedef __attribute__((ext_vector_type(8)))  __bf16 v8bf;
typedef __attribute__((ext_vector_type(8)))  float  v8f;

#define DEV __device__ __forceinline__

DEV float siluf(float x)     { return x / (1.f + __expf(-x)); }
DEV float sigm(float x)      { return 1.f / (1.f + __expf(-x)); }
DEV float softplusf(float x) { return x > 20.f ? x : __logf(1.f + __expf(x)); }

// Fragment per CDNA5 ISA 16-bit A/B-matrix 16x32 layout: lane (r,g) holds row r,
// K = {8g..8g+7} and {8g+16..8g+23} -> two contiguous 16B loads.
DEV v16bf load_frag(const bf16_t* p) {
  v8bf lo = *(const v8bf*)(p);
  v8bf hi = *(const v8bf*)(p + 16);
  v16bf f;
#pragma unroll
  for (int i = 0; i < 8; ++i) { f[i] = lo[i]; f[i + 8] = hi[i]; }
  return f;
}

// ============ direct-load WMMA GEMM: C[M,N] = A[M,K] @ B[N,K]^T (+bias) =====
// 4 waves, block tile 64x64, wave tile 32x32. For the small mamba projections.
__global__ void __launch_bounds__(128) k_gemm_bf16(
    const bf16_t* __restrict__ Am, const bf16_t* __restrict__ Bw,
    const float* __restrict__ bias, float* __restrict__ Cm,
    int M, int N, int K)
{
  const int lane = threadIdx.x & 31;
  const int wave = threadIdx.x >> 5;
  const int m0 = blockIdx.y * 64 + (wave >> 1) * 32;
  const int n0 = blockIdx.x * 64 + (wave & 1) * 32;
  if (m0 >= M || n0 >= N) return;                    // wave-uniform guard
  const int r = lane & 15, g = lane >> 4;

  const bf16_t* pa0 = Am + (size_t)(m0 + r) * K + g * 8;
  const bf16_t* pa1 = pa0 + (size_t)16 * K;
  const bf16_t* pb0 = Bw + (size_t)(n0 + r) * K + g * 8;
  const bf16_t* pb1 = pb0 + (size_t)16 * K;

  v8f acc00 = {}, acc01 = {}, acc10 = {}, acc11 = {};
  for (int kk = 0; kk < K; kk += 32) {
    v16bf a0 = load_frag(pa0 + kk);
    v16bf a1 = load_frag(pa1 + kk);
    v16bf b0 = load_frag(pb0 + kk);
    v16bf b1 = load_frag(pb1 + kk);
    if (kk + 32 < K) {
      __builtin_prefetch(pa0 + kk + 32, 0, 1);
      __builtin_prefetch(pb0 + kk + 32, 0, 1);
    }
    acc00 = __builtin_amdgcn_wmma_f32_16x16x32_bf16(false, a0, false, b0, (short)0, acc00, false, false);
    acc01 = __builtin_amdgcn_wmma_f32_16x16x32_bf16(false, a0, false, b1, (short)0, acc01, false, false);
    acc10 = __builtin_amdgcn_wmma_f32_16x16x32_bf16(false, a1, false, b0, (short)0, acc10, false, false);
    acc11 = __builtin_amdgcn_wmma_f32_16x16x32_bf16(false, a1, false, b1, (short)0, acc11, false, false);
  }
  const int mo = g * 8;                              // C/D: N=lane&15, M=vgpr+8*(lane>=16)
  const float b0v = bias ? bias[n0 + r] : 0.f;
  const float b1v = bias ? bias[n0 + 16 + r] : 0.f;
#pragma unroll
  for (int v = 0; v < 8; ++v) {
    size_t rowA = (size_t)(m0 + mo + v) * N;
    size_t rowB = (size_t)(m0 + 16 + mo + v) * N;
    Cm[rowA + n0 + r]      = acc00[v] + b0v;
    Cm[rowA + n0 + 16 + r] = acc01[v] + b1v;
    Cm[rowB + n0 + r]      = acc10[v] + b0v;
    Cm[rowB + n0 + 16 + r] = acc11[v] + b1v;
  }
}

// ====== async-LDS WMMA GEMM: block 64x128, 4 waves, wave tile 32x64 =========
// Requires M%64==0, N%128==0, K%32==0 (true for qkv / out-proj shapes).
// K-slab tiles (A: 64x32, B: 128x32 bf16) staged into double-buffered LDS by
// GLOBAL_LOAD_ASYNC_TO_LDS_B128 (ASYNCcnt), consumed via ds_load_b128 frags.
__global__ void __launch_bounds__(128) k_gemm_async(
    const bf16_t* __restrict__ Am, const bf16_t* __restrict__ Bw,
    const float* __restrict__ bias, float* __restrict__ Cm,
    int M, int N, int K)
{
  __shared__ __align__(16) bf16_t As[2][64 * 32];    // 4KB / buffer
  __shared__ __align__(16) bf16_t Bs[2][128 * 32];   // 8KB / buffer
  const int tid  = threadIdx.x;
  const int lane = tid & 31;
  const int wave = tid >> 5;
  const int m0b = blockIdx.y * 64;
  const int n0b = blockIdx.x * 128;
  const int wm0 = (wave >> 1) * 32;
  const int wn0 = (wave & 1) * 64;
  const int r = lane & 15, g = lane >> 4;

  // flat-LDS mapping: low 32 bits of a generic shared address = LDS byte offset
  uint32_t asBase[2] = { (uint32_t)(size_t)&As[0][0], (uint32_t)(size_t)&As[1][0] };
  uint32_t bsBase[2] = { (uint32_t)(size_t)&Bs[0][0], (uint32_t)(size_t)&Bs[1][0] };
  const bf16_t* Abase = Am + (size_t)m0b * K;
  const bf16_t* Bbase = Bw + (size_t)n0b * K;

  // stage copy: rows hold 32 K-values = 64B = 4 x 16B chunks
  auto issue = [&](int buf, int kk) {
#pragma unroll
    for (int i = 0; i < 2; ++i) {                    // A: 256 chunks / 128 thr
      int id = tid + i * 128;
      int row = id >> 2, ch = id & 3;
      const bf16_t* ga = Abase + (size_t)row * K + kk + ch * 8;
      uint32_t la = asBase[buf] + (uint32_t)(row * 64 + ch * 16);
      asm volatile("global_load_async_to_lds_b128 %0, %1, off"
                   :: "v"(la), "v"(ga) : "memory");
    }
#pragma unroll
    for (int i = 0; i < 4; ++i) {                    // B: 512 chunks / 128 thr
      int id = tid + i * 128;
      int row = id >> 2, ch = id & 3;
      const bf16_t* gb = Bbase + (size_t)row * K + kk + ch * 8;
      uint32_t lb = bsBase[buf] + (uint32_t)(row * 64 + ch * 16);
      asm volatile("global_load_async_to_lds_b128 %0, %1, off"
                   :: "v"(lb), "v"(gb) : "memory");
    }
  };

  v8f acc[2][4] = {};
  issue(0, 0);
  int buf = 0;
  for (int kk = 0; kk < K; kk += 32) {
    const bool more = (kk + 32) < K;
    if (more) issue(buf ^ 1, kk + 32);               // overlap next-slab DMA
    if (more) asm volatile("s_wait_asynccnt 0x6" ::: "memory");  // prev slab done
    else      asm volatile("s_wait_asynccnt 0x0" ::: "memory");
    __syncthreads();                                 // slab visible to all waves
    v16bf a[2], b[4];
#pragma unroll
    for (int mi = 0; mi < 2; ++mi)
      a[mi] = load_frag(&As[buf][(wm0 + mi * 16 + r) * 32 + g * 8]);
#pragma unroll
    for (int nj = 0; nj < 4; ++nj)
      b[nj] = load_frag(&Bs[buf][(wn0 + nj * 16 + r) * 32 + g * 8]);
#pragma unroll
    for (int mi = 0; mi < 2; ++mi)
#pragma unroll
      for (int nj = 0; nj < 4; ++nj)
        acc[mi][nj] = __builtin_amdgcn_wmma_f32_16x16x32_bf16(
            false, a[mi], false, b[nj], (short)0, acc[mi][nj], false, false);
    __syncthreads();                                 // all reads done before reuse
    buf ^= 1;
  }
  const int mo = g * 8;
#pragma unroll
  for (int mi = 0; mi < 2; ++mi) {
#pragma unroll
    for (int nj = 0; nj < 4; ++nj) {
      int ncol = n0b + wn0 + nj * 16 + r;
      float bv = bias ? bias[ncol] : 0.f;
#pragma unroll
      for (int v = 0; v < 8; ++v) {
        int mrow = m0b + wm0 + mi * 16 + mo + v;
        Cm[(size_t)mrow * N + ncol] = acc[mi][nj][v] + bv;
      }
    }
  }
}

// ======================= elementwise / helper kernels =======================
__global__ void k_cvt_bf16(const float* __restrict__ src, bf16_t* __restrict__ dst, int n) {
  int i = blockIdx.x * blockDim.x + threadIdx.x;
  if (i < n) dst[i] = (bf16_t)src[i];
}

__global__ void k_ln_w(const float* __restrict__ x, const float* __restrict__ w,
                       const float* __restrict__ b, bf16_t* __restrict__ out, int rows) {
  int ro = blockIdx.x * blockDim.x + threadIdx.x;
  if (ro >= rows) return;
  const float* p = x + (size_t)ro * 8;
  float m = 0.f;
#pragma unroll
  for (int i = 0; i < 8; ++i) m += p[i];
  m *= 0.125f;
  float var = 0.f;
#pragma unroll
  for (int i = 0; i < 8; ++i) { float d = p[i] - m; var += d * d; }
  var *= 0.125f;
  float inv = rsqrtf(var + 1e-5f);
  bf16_t* o = out + (size_t)ro * 8;
#pragma unroll
  for (int i = 0; i < 8; ++i) o[i] = (bf16_t)((p[i] - m) * inv * w[i] + b[i]);
}

__global__ void k_ln_c(const float* __restrict__ x, const float* __restrict__ w,
                       const float* __restrict__ b, bf16_t* __restrict__ out) {
  int id = blockIdx.x * blockDim.x + threadIdx.x;   // (b*64 + hw)*32 + t
  if (id >= 16 * 64 * 32) return;
  int t = id & 31, hw = (id >> 5) & 63, bb = id >> 11;
  const float* p = x + (size_t)(bb * 32 + t) * 2048 + hw;
  float m = 0.f;
#pragma unroll
  for (int c = 0; c < 32; ++c) m += p[c * 64];
  m *= (1.f / 32.f);
  float var = 0.f;
#pragma unroll
  for (int c = 0; c < 32; ++c) { float d = p[c * 64] - m; var += d * d; }
  var *= (1.f / 32.f);
  float inv = rsqrtf(var + 1e-5f);
  bf16_t* o = out + (size_t)id * 32;
#pragma unroll
  for (int c = 0; c < 32; ++c) o[c] = (bf16_t)((p[c * 64] - m) * inv * w[c] + b[c]);
}

__global__ void k_conv_silu(const float* __restrict__ xz, const float* __restrict__ cw,
                            const float* __restrict__ cb, float* __restrict__ xc,
                            int Bm, int L, int di, int k) {
  int id = blockIdx.x * blockDim.x + threadIdx.x;
  if (id >= Bm * L * di) return;
  int d = id % di, l = (id / di) % L, b = id / (di * L);
  float acc = cb[d];
  for (int j = 0; j < k; ++j) {
    int ls = l - (k - 1) + j;
    if (ls >= 0) acc += cw[d * k + j] * xz[(size_t)(b * L + ls) * (2 * di) + d];
  }
  xc[id] = siluf(acc);
}

__global__ void k_rowgemm(const float* __restrict__ X, int ldx, const float* __restrict__ Wt,
                          const float* __restrict__ bias, float* __restrict__ Y,
                          int rows, int K, int N, int act) {
  int id = blockIdx.x * blockDim.x + threadIdx.x;
  if (id >= rows * N) return;
  int n = id % N, r = id / N;
  const float* xp = X + (size_t)r * ldx;
  const float* wp = Wt + (size_t)n * K;
  float acc = bias ? bias[n] : 0.f;
  for (int k = 0; k < K; ++k) acc += xp[k] * wp[k];
  if (act == 1) acc = softplusf(acc);
  Y[(size_t)r * N + n] = acc;
}

template <int S>
__global__ void k_scan(const float* __restrict__ dbc, int edim, int dtr,
                       const float* __restrict__ dt, const float* __restrict__ xc,
                       const float* __restrict__ xz, const float* __restrict__ Alog,
                       const float* __restrict__ Dp, bf16_t* __restrict__ ybf,
                       int Bm, int L, int di) {
  int id = blockIdx.x * blockDim.x + threadIdx.x;
  if (id >= Bm * di) return;
  int d = id % di, b = id / di;
  float A[S], h[S];
#pragma unroll
  for (int s = 0; s < S; ++s) { A[s] = -__expf(Alog[d * S + s]); h[s] = 0.f; }
  float dpv = Dp[d];
  for (int l = 0; l < L; ++l) {
    size_t bl = (size_t)b * L + l;
    float dtv = dt[bl * di + d];
    float xcv = xc[bl * di + d];
    const float* bc = dbc + bl * edim + dtr;
    float y = 0.f;
#pragma unroll
    for (int s = 0; s < S; ++s) {
      h[s] = __expf(dtv * A[s]) * h[s] + dtv * bc[s] * xcv;
      y += h[s] * bc[S + s];
    }
    float zv = xz[bl * (2 * di) + di + d];
    y = (y + dpv * xcv) * siluf(zv);
    ybf[bl * di + d] = (bf16_t)y;
  }
}

__global__ void k_temporal(const float* __restrict__ x,
                           const float* __restrict__ ltw, const float* __restrict__ ltb,
                           const float* __restrict__ Win, const float* __restrict__ cw,
                           const float* __restrict__ cb, const float* __restrict__ Wx,
                           const float* __restrict__ Wdt, const float* __restrict__ bdt,
                           const float* __restrict__ Alog, const float* __restrict__ Dpp,
                           const float* __restrict__ Wout, float* __restrict__ tf) {
  int q = blockIdx.x * blockDim.x + threadIdx.x;
  if (q >= 16 * 2048) return;
  int b = q >> 11, chw = q & 2047;
  const float* xp = x + (size_t)b * 32 * 2048 + chw;
  float v[32];
  float m = 0.f;
#pragma unroll
  for (int t = 0; t < 32; ++t) { v[t] = xp[(size_t)t * 2048]; m += v[t]; }
  m *= (1.f / 32.f);
  float var = 0.f;
#pragma unroll
  for (int t = 0; t < 32; ++t) { float d = v[t] - m; var += d * d; }
  var *= (1.f / 32.f);
  float inv = rsqrtf(var + 1e-5f);
#pragma unroll
  for (int t = 0; t < 32; ++t) v[t] = (v[t] - m) * inv * ltw[t] + ltb[t];

  float wi0 = Win[0], wi1 = Win[1];
  float c0 = cw[0], c1 = cw[1], c2 = cw[2], cbv = cb[0];
  float wx0 = Wx[0], wdt = Wdt[0], bdtv = bdt[0];
  float dpv = Dpp[0], wo = Wout[0];
  float A[12], Bc[12], Cc[12], h[12];
#pragma unroll
  for (int s = 0; s < 12; ++s) {
    A[s] = -__expf(Alog[s]); Bc[s] = Wx[1 + s]; Cc[s] = Wx[13 + s]; h[s] = 0.f;
  }
  for (int t = 0; t < 32; ++t) {
    float x0  = wi0 * v[t];
    float xm1 = (t >= 1) ? wi0 * v[t - 1] : 0.f;
    float xm2 = (t >= 2) ? wi0 * v[t - 2] : 0.f;
    float xc = siluf(c0 * xm2 + c1 * xm1 + c2 * x0 + cbv);
    float dt = softplusf(xc * wx0 * wdt + bdtv);
    float y = 0.f;
#pragma unroll
    for (int s = 0; s < 12; ++s) {
      h[s] = __expf(dt * A[s]) * h[s] + dt * (xc * Bc[s]) * xc;
      y += h[s] * (xc * Cc[s]);
    }
    float z = wi1 * v[t];
    y = (y + dpv * xc) * siluf(z);
    tf[(size_t)(b * 32 + t) * 2048 + chw] = y * wo;
  }
}

__global__ void k_cf_tr(const float* __restrict__ cfb, float* __restrict__ cf) {
  int id = blockIdx.x * blockDim.x + threadIdx.x;
  if (id >= 1048576) return;
  int hw = id & 63, c = (id >> 6) & 31, t = (id >> 11) & 31, b = id >> 16;
  cf[id] = cfb[(((size_t)(b * 64 + hw)) * 32 + t) * 32 + c];
}

__global__ void k_combine(const float* __restrict__ sf, const float* __restrict__ cf,
                          const float* __restrict__ tf, bf16_t* __restrict__ comb) {
  int id = blockIdx.x * blockDim.x + threadIdx.x;
  if (id >= 1536 * 2048) return;
  int e = id & 2047, row = id >> 11, br = row % 3, n = row / 3;
  const float* src = (br == 0) ? sf : (br == 1) ? cf : tf;
  comb[id] = (bf16_t)src[(size_t)n * 2048 + e];
}

__global__ void __launch_bounds__(32) k_attn(const float* __restrict__ qkv,
                                             bf16_t* __restrict__ obf) {
  const int n = blockIdx.x >> 2;
  const int hh = blockIdx.x & 3;
  const int lane = threadIdx.x;
  const float* base = qkv + (size_t)n * 3 * 6144 + hh * 512;
  float s[3][3];
#pragma unroll
  for (int i = 0; i < 3; ++i) {
    const float* qi = base + i * 6144;
#pragma unroll
    for (int j = 0; j < 3; ++j) {
      const float* kj = base + j * 6144 + 2048;
      float p = 0.f;
#pragma unroll
      for (int e = 0; e < 16; ++e) p += qi[lane * 16 + e] * kj[lane * 16 + e];
      for (int mm = 16; mm > 0; mm >>= 1) p += __shfl_xor(p, mm, 32);
      s[i][j] = p * 0.04419417382415922f;  // 512^-0.5
    }
  }
  float att[3][3];
#pragma unroll
  for (int i = 0; i < 3; ++i) {
    float mx = fmaxf(s[i][0], fmaxf(s[i][1], s[i][2]));
    float e0 = __expf(s[i][0] - mx), e1 = __expf(s[i][1] - mx), e2 = __expf(s[i][2] - mx);
    float is = 1.f / (e0 + e1 + e2);
    att[i][0] = e0 * is; att[i][1] = e1 * is; att[i][2] = e2 * is;
  }
  const float* v0 = base + 4096;
  const float* v1 = base + 6144 + 4096;
  const float* v2 = base + 12288 + 4096;
#pragma unroll
  for (int i = 0; i < 3; ++i) {
#pragma unroll
    for (int e = 0; e < 16; ++e) {
      int idx = lane * 16 + e;
      float o = att[i][0] * v0[idx] + att[i][1] * v1[idx] + att[i][2] * v2[idx];
      obf[((size_t)n * 3 + i) * 2048 + hh * 512 + idx] = (bf16_t)o;
    }
  }
}

__global__ void k_means(const float* __restrict__ sf, const float* __restrict__ cf,
                        const float* __restrict__ tf, float* __restrict__ means) {
  int id = blockIdx.x * blockDim.x + threadIdx.x;
  if (id >= 512 * 96) return;
  int f = id % 96, n = id / 96;
  const float* src = (f < 32) ? sf : (f < 64) ? cf : tf;
  const float* p = src + (size_t)n * 2048 + (f & 31) * 64;
  float m = 0.f;
#pragma unroll
  for (int i = 0; i < 64; ++i) m += p[i];
  means[id] = m * (1.f / 64.f);
}

__global__ void k_gate(const float* __restrict__ means, const float* __restrict__ gw,
                       const float* __restrict__ gbv, float* __restrict__ g) {
  int id = blockIdx.x * blockDim.x + threadIdx.x;
  if (id >= 512 * 32) return;
  int c = id & 31, n = id >> 5;
  const float* mp = means + (size_t)n * 96;
  const float* wp = gw + (size_t)c * 96;
  float acc = gbv[c];
  for (int f = 0; f < 96; ++f) acc += mp[f] * wp[f];
  g[id] = sigm(acc);
}

__global__ void k_final(const float* __restrict__ attno, const float* __restrict__ g,
                        float* __restrict__ out) {
  int id = blockIdx.x * blockDim.x + threadIdx.x;
  if (id >= 512 * 2048) return;
  int e = id & 2047, n = id >> 11;
  float a0 = attno[((size_t)n * 3 + 0) * 2048 + e];
  float a1 = attno[((size_t)n * 3 + 1) * 2048 + e];
  float gv = g[n * 32 + (e >> 6)];
  out[id] = gv * a0 + (1.f - gv) * a1;
}

// ================================ host side =================================
static inline int cdiv(int a, int b) { return (a + b - 1) / b; }

extern "C" void kernel_launch(void* const* d_in, const int* in_sizes, int n_in,
                              void* d_out, int out_size, void* d_ws, size_t ws_size,
                              hipStream_t stream)
{
  (void)in_sizes; (void)n_in; (void)out_size; (void)ws_size;
  const float* x      = (const float*)d_in[0];
  const float* spWin  = (const float*)d_in[1];
  const float* spCw   = (const float*)d_in[2];
  const float* spCb   = (const float*)d_in[3];
  const float* spWx   = (const float*)d_in[4];
  const float* spWdt  = (const float*)d_in[5];
  const float* spBdt  = (const float*)d_in[6];
  const float* spAlog = (const float*)d_in[7];
  const float* spDp   = (const float*)d_in[8];
  const float* spWout = (const float*)d_in[9];
  const float* scWin  = (const float*)d_in[10];
  const float* scCw   = (const float*)d_in[11];
  const float* scCb   = (const float*)d_in[12];
  const float* scWx   = (const float*)d_in[13];
  const float* scWdt  = (const float*)d_in[14];
  const float* scBdt  = (const float*)d_in[15];
  const float* scAlog = (const float*)d_in[16];
  const float* scDp   = (const float*)d_in[17];
  const float* scWout = (const float*)d_in[18];
  const float* tpWin  = (const float*)d_in[19];
  const float* tpCw   = (const float*)d_in[20];
  const float* tpCb   = (const float*)d_in[21];
  const float* tpWx   = (const float*)d_in[22];
  const float* tpWdt  = (const float*)d_in[23];
  const float* tpBdt  = (const float*)d_in[24];
  const float* tpAlog = (const float*)d_in[25];
  const float* tpDp   = (const float*)d_in[26];
  const float* tpWout = (const float*)d_in[27];
  const float* lsw = (const float*)d_in[28];
  const float* lsb = (const float*)d_in[29];
  const float* lcw = (const float*)d_in[30];
  const float* lcb = (const float*)d_in[31];
  const float* ltw = (const float*)d_in[32];
  const float* ltb = (const float*)d_in[33];
  const float* Wi  = (const float*)d_in[34];
  const float* bi  = (const float*)d_in[35];
  const float* Wo  = (const float*)d_in[36];
  const float* bo  = (const float*)d_in[37];
  const float* gw  = (const float*)d_in[38];
  const float* gbv = (const float*)d_in[39];

  char* ws = (char*)d_ws;
  size_t off = 0;
  auto alloc = [&](size_t bytes) -> char* {
    char* p = ws + off; off += (bytes + 255) & ~(size_t)255; return p;
  };
  float*  sf      = (float*)alloc(16384ull * 64 * 4);
  float*  cfb     = (float*)alloc(32768ull * 32 * 4);
  float*  cf      = (float*)alloc(1048576ull * 4);
  float*  tf      = (float*)alloc(1048576ull * 4);
  bf16_t* comb    = (bf16_t*)alloc(1536ull * 2048 * 2);
  bf16_t* WiB     = (bf16_t*)alloc(6144ull * 2048 * 2);
  bf16_t* WoB     = (bf16_t*)alloc(2048ull * 2048 * 2);
  bf16_t* spWinB  = (bf16_t*)alloc(256 * 64 * 2);
  bf16_t* spWoutB = (bf16_t*)alloc(64 * 128 * 2);
  bf16_t* scWinB  = (bf16_t*)alloc(64 * 32 * 2);
  bf16_t* scWoutB = (bf16_t*)alloc(32 * 32 * 2);
  bf16_t* obf     = (bf16_t*)alloc(1536ull * 2048 * 2);
  float*  attno   = (float*)alloc(1536ull * 2048 * 4);
  float*  meansb  = (float*)alloc(512 * 96 * 4);
  float*  gbuf    = (float*)alloc(512 * 32 * 4);
  char*   regA    = alloc(43ull * 1024 * 1024);

  k_cvt_bf16<<<cdiv(6144 * 2048, 256), 256, 0, stream>>>(Wi, WiB, 6144 * 2048);
  k_cvt_bf16<<<cdiv(2048 * 2048, 256), 256, 0, stream>>>(Wo, WoB, 2048 * 2048);
  k_cvt_bf16<<<cdiv(256 * 64, 256), 256, 0, stream>>>(spWin, spWinB, 256 * 64);
  k_cvt_bf16<<<cdiv(64 * 128, 256), 256, 0, stream>>>(spWout, spWoutB, 64 * 128);
  k_cvt_bf16<<<cdiv(64 * 32, 256), 256, 0, stream>>>(scWin, scWinB, 64 * 32);
  k_cvt_bf16<<<cdiv(32 * 32, 256), 256, 0, stream>>>(scWout, scWoutB, 32 * 32);

  // -------- spatial branch: Bm=16, L=1024, d_model=64, d_inner=128, S=16 ----
  {
    bf16_t* sfn = (bf16_t*)(regA + 0);
    float*  xz1 = (float*)(regA + 2097152);
    float*  xc1 = (float*)(regA + 18874368);
    float*  db1 = (float*)(regA + 27262976);
    float*  dt1 = (float*)(regA + 29622272);
    bf16_t* yb1 = (bf16_t*)(regA + 38010880);
    k_ln_w<<<cdiv(131072, 256), 256, 0, stream>>>(x, lsw, lsb, sfn, 131072);
    k_gemm_bf16<<<dim3(4, 256), 128, 0, stream>>>(sfn, spWinB, nullptr, xz1, 16384, 256, 64);
    k_conv_silu<<<cdiv(16 * 1024 * 128, 256), 256, 0, stream>>>(xz1, spCw, spCb, xc1, 16, 1024, 128, 4);
    k_rowgemm<<<cdiv(16384 * 36, 256), 256, 0, stream>>>(xc1, 128, spWx, nullptr, db1, 16384, 128, 36, 0);
    k_rowgemm<<<cdiv(16384 * 128, 256), 256, 0, stream>>>(db1, 36, spWdt, spBdt, dt1, 16384, 4, 128, 1);
    k_scan<16><<<cdiv(2048, 256), 256, 0, stream>>>(db1, 36, 4, dt1, xc1, xz1, spAlog, spDp, yb1, 16, 1024, 128);
    k_gemm_bf16<<<dim3(1, 256), 128, 0, stream>>>(yb1, spWoutB, nullptr, sf, 16384, 64, 128);
  }
  // -------- spectral branch: Bm=1024, L=32, d_model=32, d_inner=32, S=8 -----
  {
    bf16_t* cfn = (bf16_t*)(regA + 0);
    float*  xz2 = (float*)(regA + 2097152);
    float*  xc2 = (float*)(regA + 18874368);
    float*  db2 = (float*)(regA + 27262976);
    float*  dt2 = (float*)(regA + 29622272);
    bf16_t* yb2 = (bf16_t*)(regA + 38010880);
    k_ln_c<<<cdiv(32768, 256), 256, 0, stream>>>(x, lcw, lcb, cfn);
    k_gemm_bf16<<<dim3(1, 512), 128, 0, stream>>>(cfn, scWinB, nullptr, xz2, 32768, 64, 32);
    k_conv_silu<<<cdiv(1024 * 32 * 32, 256), 256, 0, stream>>>(xz2, scCw, scCb, xc2, 1024, 32, 32, 3);
    k_rowgemm<<<cdiv(32768 * 18, 256), 256, 0, stream>>>(xc2, 32, scWx, nullptr, db2, 32768, 32, 18, 0);
    k_rowgemm<<<cdiv(32768 * 32, 256), 256, 0, stream>>>(db2, 18, scWdt, scBdt, dt2, 32768, 2, 32, 1);
    k_scan<8><<<cdiv(32768, 256), 256, 0, stream>>>(db2, 18, 2, dt2, xc2, xz2, scAlog, scDp, yb2, 1024, 32, 32);
    k_gemm_bf16<<<dim3(1, 512), 128, 0, stream>>>(yb2, scWoutB, nullptr, cfb, 32768, 32, 32);
    k_cf_tr<<<cdiv(1048576, 256), 256, 0, stream>>>(cfb, cf);
  }
  // -------- temporal branch (fully fused) -----------------------------------
  k_temporal<<<cdiv(32768, 256), 256, 0, stream>>>(x, ltw, ltb, tpWin, tpCw, tpCb,
                                                   tpWx, tpWdt, tpBdt, tpAlog, tpDp, tpWout, tf);
  // -------- cross attention: async-LDS WMMA GEMMs ----------------------------
  k_combine<<<cdiv(1536 * 2048, 256), 256, 0, stream>>>(sf, cf, tf, comb);
  float* qkv = (float*)(regA + 0);  // 37.7MB, reuses branch-temp region
  k_gemm_async<<<dim3(48, 24), 128, 0, stream>>>(comb, WiB, bi, qkv, 1536, 6144, 2048);
  k_attn<<<2048, 32, 0, stream>>>(qkv, obf);
  k_gemm_async<<<dim3(16, 24), 128, 0, stream>>>(obf, WoB, bo, attno, 1536, 2048, 2048);
  // -------- gated fusion -----------------------------------------------------
  k_means<<<cdiv(512 * 96, 256), 256, 0, stream>>>(sf, cf, tf, meansb);
  k_gate<<<cdiv(512 * 32, 256), 256, 0, stream>>>(meansb, gw, gbv, gbuf);
  k_final<<<cdiv(512 * 2048, 256), 256, 0, stream>>>(attno, gbuf, (float*)d_out);
}
